// MoE_78709570666646
// MI455X (gfx1250) — matrix operand order repaired and checked by reading, not compiled, not run on hardware
//
#include <hip/hip_runtime.h>
#include <hip/hip_bf16.h>

typedef __attribute__((ext_vector_type(16))) __bf16 v16bf;
typedef __attribute__((ext_vector_type(8)))  __bf16 v8bf;
typedef __attribute__((ext_vector_type(8)))  float  v8f;
typedef __attribute__((ext_vector_type(4)))  unsigned int v4u;
typedef __attribute__((ext_vector_type(8)))  int v8i;
typedef __attribute__((ext_vector_type(4)))  int v4i;

#define N_TOK 8192
#define H_DIM 1024
#define I_DIM 4096
#define N_EXP 8

// ---------------- workspace layout (bytes) ----------------
// xb      : N*H bf16            =  16 MB  @ 0
// w1t     : E*I*H bf16 (transp) =  64 MB  @ 16777216
// wot     : H*I bf16  (transp)  =   8 MB  @ 83886080
// interb  : N*I bf16            =  64 MB  @ 92274688
// y       : N*H f32             =  32 MB  @ 159383552
// counts  : E int                         @ 192937984
// lists   : E*N int             = 256 KB  @ 192938240
// total ~184.3 MB

// -------------------------------------------------------------------------
// TDM 2D tile load: global [tile_rows][tile_k] bf16 (row stride row_stride
// elems) -> LDS at lds_addr, rows packed contiguously ([tile_rows][tile_k]).
// D# per CDNA5 ISA 8.3/8.4. Issued by one wave; tracked by TENSORcnt.
__device__ inline void tdm_load_2d(unsigned lds_addr, const void* gptr,
                                   unsigned tile_k, unsigned tile_rows,
                                   unsigned row_stride_elems) {
    unsigned long long ga = (unsigned long long)(size_t)gptr;
    v4u g0;
    g0[0] = 1u;                                               // count=1, user D#
    g0[1] = lds_addr;                                         // LDS byte address
    g0[2] = (unsigned)ga;                                     // global_addr[31:0]
    g0[3] = ((unsigned)(ga >> 32) & 0x01FFFFFFu) | (2u << 30); // ga[56:32] | type=2
    v8i g1;
    g1[0] = (int)(1u << 16);                                  // data_size=1 (2 bytes)
    g1[1] = (int)((tile_k & 0xFFFFu) << 16);                  // tensor_dim0[15:0]
    g1[2] = (int)(((tile_k >> 16) & 0xFFFFu) |
                  ((tile_rows & 0xFFFFu) << 16));             // td0[31:16] | td1[15:0]
    g1[3] = (int)(((tile_rows >> 16) & 0xFFFFu) |
                  (tile_k << 16));                            // td1[31:16] | tile_dim0
    g1[4] = (int)tile_rows;                                   // tile_dim1 (tile_dim2=0)
    g1[5] = (int)row_stride_elems;                            // tensor_dim0_stride lo
    g1[6] = 0;
    g1[7] = 0;
    v4i z4 = {0, 0, 0, 0};
#if __has_include(<hip/amd_detail/amd_gfx1250_TDM.h>)
    v8i z8 = {0, 0, 0, 0, 0, 0, 0, 0};
    __builtin_amdgcn_tensor_load_to_lds(g0, g1, z4, z4, z8, 0);   // clang-23 form
#else
    __builtin_amdgcn_tensor_load_to_lds(g0, g1, z4, z4, 0);       // ROCm 7.2 form
#endif
}

__device__ inline unsigned lds_off(const void* p) {
    // LDS aperture: flat addr low 32 bits are the LDS byte offset (ISA 10.2)
    return (unsigned)(size_t)p;
}

// -------------------------------------------------------------------------
__global__ void zero_counts_kernel(int* counts) {
    if (threadIdx.x < N_EXP) counts[threadIdx.x] = 0;
}

// Tiled transpose + f32->bf16: src [z][R][C] f32 -> dst [z][C][R] bf16.
__global__ void __launch_bounds__(256) transpose_convert_kernel(
        const float* __restrict__ src, __bf16* __restrict__ dst, int R, int C) {
    const size_t zoff = (size_t)blockIdx.z * R * C;
    const int r0 = blockIdx.y * 64, c0 = blockIdx.x * 64;
    __shared__ __align__(16) __bf16 tile[64][72];   // 144B row stride, 16B aligned
    const int tid = threadIdx.x;
#pragma unroll
    for (int p = 0; p < 4; ++p) {
        int row = (tid >> 4) + p * 16;
        int col = (tid & 15) * 4;
        const float4 v = *(const float4*)&src[zoff + (size_t)(r0 + row) * C + c0 + col];
        tile[col + 0][row] = (__bf16)v.x;
        tile[col + 1][row] = (__bf16)v.y;
        tile[col + 2][row] = (__bf16)v.z;
        tile[col + 3][row] = (__bf16)v.w;
    }
    __syncthreads();
#pragma unroll
    for (int p = 0; p < 2; ++p) {
        int chunk = tid + p * 256;
        int orow = chunk >> 3, cc = (chunk & 7) * 8;
        *(v8bf*)&dst[zoff + (size_t)(c0 + orow) * R + r0 + cc] =
            *(const v8bf*)&tile[orow][cc];
    }
}

// Router: one wave per token. Wr (8x1024 f32 = 32KB) cached in LDS.
__global__ void __launch_bounds__(256) router_kernel(const float* __restrict__ x,
                                                     const float* __restrict__ Wr,
                                                     __bf16* __restrict__ xb,
                                                     int* __restrict__ counts,
                                                     int* __restrict__ lists) {
    __shared__ float wr[N_EXP * H_DIM];
    const int tid = threadIdx.x;
    for (int i = tid; i < N_EXP * H_DIM; i += 256) wr[i] = Wr[i];
    __syncthreads();

    const int wave = tid >> 5, lane = tid & 31;
    const int t = blockIdx.x * 8 + wave;

    float acc[N_EXP];
#pragma unroll
    for (int e = 0; e < N_EXP; ++e) acc[e] = 0.0f;

    for (int i = 0; i < H_DIM / 32; ++i) {
        int h = i * 32 + lane;
        float xv = x[(size_t)t * H_DIM + h];
        xb[(size_t)t * H_DIM + h] = (__bf16)xv;
#pragma unroll
        for (int e = 0; e < N_EXP; ++e) acc[e] += xv * wr[e * H_DIM + h];
    }
#pragma unroll
    for (int off = 16; off > 0; off >>= 1)
#pragma unroll
        for (int e = 0; e < N_EXP; ++e) acc[e] += __shfl_xor(acc[e], off, 32);

    if (lane == 0) {
        int best = 0; float bv = acc[0];
#pragma unroll
        for (int e = 1; e < N_EXP; ++e)
            if (acc[e] > bv) { bv = acc[e]; best = e; }   // first-max wins, like argmax
        int pos = atomicAdd(&counts[best], 1);
        lists[best * N_TOK + pos] = t;
    }
}

// -------------------------------------------------------------------------
// WMMA fragment loaders (ISA 7.12.2 layouts). LDS tiles are [rows][64] bf16.
__device__ inline v16bf frag_a(const __bf16* base, int row0, int ks, int lane) {
    int r = lane & 15, h = lane >> 4;
    union { v16bf v; v8bf p[2]; } f;
    const __bf16* ptr = base + (row0 + r) * 64 + ks;
    f.p[0] = *(const v8bf*)(ptr + h * 8);
    f.p[1] = *(const v8bf*)(ptr + 16 + h * 8);
    return f.v;
}
__device__ inline v16bf frag_b(const __bf16* base, int col0, int ks, int lane) {
    int r = lane & 15, h = lane >> 4;
    union { v16bf v; v8bf p[2]; } f;
    const __bf16* ptr = base + (col0 + r) * 64 + ks;
    f.p[0] = *(const v8bf*)(ptr + h * 8);
    f.p[1] = *(const v8bf*)(ptr + 16 + h * 8);
    return f.v;
}

__device__ inline float gelu_exact(float v) {
    return 0.5f * v * (1.0f + erff(v * 0.70710678118654752f));
}

// 8-WMMA burst on preloaded fragments: all DS loads issued before any WMMA
// so waits interleave at descending thresholds instead of full drains.
#define WMMA_STEP(abuf, bbuf, ks)                                              \
    {                                                                          \
        v16bf a0 = frag_a((abuf), mw,      (ks), lane);                        \
        v16bf a1 = frag_a((abuf), mw + 16, (ks), lane);                        \
        v16bf b0 = frag_b((bbuf), nw,      (ks), lane);                        \
        v16bf b1v = frag_b((bbuf), nw + 16, (ks), lane);                       \
        v16bf b2 = frag_b((bbuf), nw + 32, (ks), lane);                        \
        v16bf b3 = frag_b((bbuf), nw + 48, (ks), lane);                        \
        acc[0][0] = __builtin_amdgcn_wmma_f32_16x16x32_bf16(                   \
            false, a0, false, b0, (short)0, acc[0][0], false, false);          \
        acc[1][0] = __builtin_amdgcn_wmma_f32_16x16x32_bf16(                   \
            false, a1, false, b0, (short)0, acc[1][0], false, false);          \
        acc[0][1] = __builtin_amdgcn_wmma_f32_16x16x32_bf16(                   \
            false, a0, false, b1v, (short)0, acc[0][1], false, false);         \
        acc[1][1] = __builtin_amdgcn_wmma_f32_16x16x32_bf16(                   \
            false, a1, false, b1v, (short)0, acc[1][1], false, false);         \
        acc[0][2] = __builtin_amdgcn_wmma_f32_16x16x32_bf16(                   \
            false, a0, false, b2, (short)0, acc[0][2], false, false);          \
        acc[1][2] = __builtin_amdgcn_wmma_f32_16x16x32_bf16(                   \
            false, a1, false, b2, (short)0, acc[1][2], false, false);          \
        acc[0][3] = __builtin_amdgcn_wmma_f32_16x16x32_bf16(                   \
            false, a0, false, b3, (short)0, acc[0][3], false, false);          \
        acc[1][3] = __builtin_amdgcn_wmma_f32_16x16x32_bf16(                   \
            false, a1, false, b3, (short)0, acc[1][3], false, false);          \
    }

// -------------------------------------------------------------------------
// Grouped up-projection: per expert, inter = gelu(X_e @ W1[e] + b1[e])
// grid = (I/128, N/128, E); 8 waves; tile 128x128, K-chunk 64.
// A: cooperative gathered loads (pointers hoisted). B: TDM double-buffered.
__global__ void __launch_bounds__(256) moe_up_kernel(const __bf16* __restrict__ xb,
                                                     const __bf16* __restrict__ w1t,
                                                     const float* __restrict__ b1,
                                                     const int* __restrict__ counts,
                                                     const int* __restrict__ lists,
                                                     __bf16* __restrict__ interb) {
    const int e = blockIdx.z;
    const int nE = counts[e];
    const int m0 = blockIdx.y * 128;
    if (m0 >= nE) return;
    const int nb = blockIdx.x * 128;

    __shared__ __align__(16) __bf16 As[128][64];
    __shared__ __align__(16) __bf16 Bs[2][128][64];
    __shared__ int toks[128];

    const int tid = threadIdx.x;
    for (int i = tid; i < 128; i += 256) {
        int gr = m0 + i; if (gr > nE - 1) gr = nE - 1;
        toks[i] = lists[e * N_TOK + gr];
    }
    __syncthreads();

    const __bf16* wE = w1t + (size_t)e * I_DIM * H_DIM;   // [I][H], K contiguous
    const __bf16* bsrc = wE + (size_t)nb * H_DIM;         // tile row base
    const unsigned bl0 = lds_off(&Bs[0][0][0]), bl1 = lds_off(&Bs[1][0][0]);

    v8f acc[2][4];
#pragma unroll
    for (int i = 0; i < 2; ++i)
#pragma unroll
        for (int j = 0; j < 4; ++j) acc[i][j] = {};

    const int wave = tid >> 5, lane = tid & 31;
    const int mw = (wave & 3) * 32, nw = (wave >> 2) * 64;
    const int lr = tid >> 3, lc = (tid & 7) * 8;

    // hoisted gathered-row pointers for the A tile (bumped by 64 elems/iter)
    const __bf16* ap[4];
#pragma unroll
    for (int p = 0; p < 4; ++p)
        ap[p] = xb + (size_t)toks[lr + p * 32] * H_DIM + lc;

    if (tid < 32) tdm_load_2d(bl0, bsrc, 64, 128, H_DIM);  // tile kb=0 -> buf0

    int cur = 0;
    for (int kb = 0; kb < H_DIM; kb += 64) {
#pragma unroll
        for (int p = 0; p < 4; ++p) {
            *(v8bf*)&As[lr + p * 32][lc] = *(const v8bf*)ap[p];
            ap[p] += 64;
        }
        if (kb + 64 < H_DIM)   // hint next A chunk (global_prefetch_b8)
            __builtin_prefetch(ap[0], 0, 0);
        if (tid < 32) __builtin_amdgcn_s_wait_tensorcnt(0);
        __syncthreads();   // A published + B tile kb ready in Bs[cur]
        if (kb + 64 < H_DIM && tid < 32)
            tdm_load_2d(cur ? bl0 : bl1, bsrc + kb + 64, 64, 128, H_DIM);

        const __bf16* bbuf = &Bs[cur][0][0];
        const __bf16* abuf = &As[0][0];
        WMMA_STEP(abuf, bbuf, 0)
        WMMA_STEP(abuf, bbuf, 32)
        __syncthreads();
        cur ^= 1;
    }

    const float* b1e = b1 + e * I_DIM;
    const int col = lane & 15, rb = (lane >> 4) * 8;
#pragma unroll
    for (int mi = 0; mi < 2; ++mi)
#pragma unroll
        for (int nj = 0; nj < 4; ++nj)
#pragma unroll
            for (int i = 0; i < 8; ++i) {
                int m = mw + mi * 16 + rb + i;
                if (m0 + m < nE) {
                    int n = nb + nw + nj * 16 + col;
                    float v = acc[mi][nj][i] + b1e[n];
                    interb[(size_t)toks[m] * I_DIM + n] = (__bf16)gelu_exact(v);
                }
            }
}

// -------------------------------------------------------------------------
// Down-projection (shared Wo): y = inter @ Wo + bo + x
// grid = (H/128, N/128); tile 128x128, K = I = 4096.
// Both A and B tiles streamed by TDM, double-buffered: inner loop has no
// per-wave VMEM at all — only ds_load_b128 + WMMA.
__global__ void __launch_bounds__(256) moe_down_kernel(const __bf16* __restrict__ interb,
                                                       const __bf16* __restrict__ wot,
                                                       const float* __restrict__ bo,
                                                       const float* __restrict__ x,
                                                       float* __restrict__ y) {
    const int m0 = blockIdx.y * 128;
    const int nb = blockIdx.x * 128;

    __shared__ __align__(16) __bf16 As[2][128][64];
    __shared__ __align__(16) __bf16 Bs[2][128][64];

    const int tid = threadIdx.x;
    v8f acc[2][4];
#pragma unroll
    for (int i = 0; i < 2; ++i)
#pragma unroll
        for (int j = 0; j < 4; ++j) acc[i][j] = {};

    const int wave = tid >> 5, lane = tid & 31;
    const int mw = (wave & 3) * 32, nw = (wave >> 2) * 64;

    const __bf16* asrc = interb + (size_t)m0 * I_DIM;   // [128 tokens][I]
    const __bf16* bsrc = wot + (size_t)nb * I_DIM;      // [128 h-cols][I]
    const unsigned al0 = lds_off(&As[0][0][0]), al1 = lds_off(&As[1][0][0]);
    const unsigned bl0 = lds_off(&Bs[0][0][0]), bl1 = lds_off(&Bs[1][0][0]);

    if (tid < 32) {
        tdm_load_2d(al0, asrc, 64, 128, I_DIM);
        tdm_load_2d(bl0, bsrc, 64, 128, I_DIM);
    }

    int cur = 0;
    for (int kb = 0; kb < I_DIM; kb += 64) {
        if (tid < 32) __builtin_amdgcn_s_wait_tensorcnt(0);
        __syncthreads();   // tiles kb ready in As[cur]/Bs[cur]
        if (kb + 64 < I_DIM && tid < 32) {
            tdm_load_2d(cur ? al0 : al1, asrc + kb + 64, 64, 128, I_DIM);
            tdm_load_2d(cur ? bl0 : bl1, bsrc + kb + 64, 64, 128, I_DIM);
        }

        const __bf16* abuf = &As[cur][0][0];
        const __bf16* bbuf = &Bs[cur][0][0];
        WMMA_STEP(abuf, bbuf, 0)
        WMMA_STEP(abuf, bbuf, 32)
        __syncthreads();
        cur ^= 1;
    }

    const int col = lane & 15, rb = (lane >> 4) * 8;
#pragma unroll
    for (int mi = 0; mi < 2; ++mi)
#pragma unroll
        for (int nj = 0; nj < 4; ++nj)
#pragma unroll
            for (int i = 0; i < 8; ++i) {
                int t = m0 + mw + mi * 16 + rb + i;
                int n = nb + nw + nj * 16 + col;
                y[(size_t)t * H_DIM + n] = acc[mi][nj][i] + bo[n] + x[(size_t)t * H_DIM + n];
            }
}

// -------------------------------------------------------------------------
// LayerNorm over H=1024, one token per 256-thread block.
__global__ void __launch_bounds__(256) ln_kernel(const float* __restrict__ y,
                                                 const float* __restrict__ gamma,
                                                 const float* __restrict__ beta,
                                                 float* __restrict__ out) {
    const int t = blockIdx.x;
    const float* yr = y + (size_t)t * H_DIM;
    float v[4], s = 0.0f, s2 = 0.0f;
#pragma unroll
    for (int i = 0; i < 4; ++i) {
        v[i] = yr[threadIdx.x + i * 256];
        s += v[i];
        s2 += v[i] * v[i];
    }
#pragma unroll
    for (int off = 16; off > 0; off >>= 1) {
        s  += __shfl_xor(s, off, 32);
        s2 += __shfl_xor(s2, off, 32);
    }
    __shared__ float ws1[8], ws2[8];
    int wv = threadIdx.x >> 5, ln = threadIdx.x & 31;
    if (ln == 0) { ws1[wv] = s; ws2[wv] = s2; }
    __syncthreads();
    s = 0.0f; s2 = 0.0f;
#pragma unroll
    for (int i = 0; i < 8; ++i) { s += ws1[i]; s2 += ws2[i]; }
    float mu = s * (1.0f / H_DIM);
    float var = s2 * (1.0f / H_DIM) - mu * mu;
    float inv = rsqrtf(var + 1e-12f);
#pragma unroll
    for (int i = 0; i < 4; ++i) {
        int h = threadIdx.x + i * 256;
        out[(size_t)t * H_DIM + h] = (v[i] - mu) * inv * gamma[h] + beta[h];
    }
}

// -------------------------------------------------------------------------
extern "C" void kernel_launch(void* const* d_in, const int* in_sizes, int n_in,
                              void* d_out, int out_size, void* d_ws, size_t ws_size,
                              hipStream_t stream) {
    const float* x     = (const float*)d_in[0];
    const float* Wr    = (const float*)d_in[1];
    const float* W1    = (const float*)d_in[2];
    const float* b1    = (const float*)d_in[3];
    const float* Wo    = (const float*)d_in[4];
    const float* bo    = (const float*)d_in[5];
    const float* gamma = (const float*)d_in[6];
    const float* beta  = (const float*)d_in[7];

    char* ws = (char*)d_ws;
    __bf16* xb     = (__bf16*)(ws + 0);
    __bf16* w1t    = (__bf16*)(ws + 16777216);   // [E][I][H]
    __bf16* wot    = (__bf16*)(ws + 83886080);   // [H][I]
    __bf16* interb = (__bf16*)(ws + 92274688);
    float*  y      = (float*)(ws + 159383552);
    int*    counts = (int*)(ws + 192937984);
    int*    lists  = (int*)(ws + 192938240);

    zero_counts_kernel<<<1, 32, 0, stream>>>(counts);
    // W1 [E][H][I] -> w1t [E][I][H]
    transpose_convert_kernel<<<dim3(I_DIM / 64, H_DIM / 64, N_EXP), 256, 0, stream>>>(
        W1, w1t, H_DIM, I_DIM);
    // Wo [I][H] -> wot [H][I]
    transpose_convert_kernel<<<dim3(H_DIM / 64, I_DIM / 64, 1), 256, 0, stream>>>(
        Wo, wot, I_DIM, H_DIM);
    router_kernel<<<N_TOK / 8, 256, 0, stream>>>(x, Wr, xb, counts, lists);
    moe_up_kernel<<<dim3(I_DIM / 128, N_TOK / 128, N_EXP), 256, 0, stream>>>(
        xb, w1t, b1, counts, lists, interb);
    moe_down_kernel<<<dim3(H_DIM / 128, N_TOK / 128), 256, 0, stream>>>(
        interb, wot, bo, x, y);
    ln_kernel<<<N_TOK, 256, 0, stream>>>(y, gamma, beta, (float*)d_out);
}
